// NeSyBase_36575941493336
// MI455X (gfx1250) — matrix operand order repaired and checked by reading, not compiled, not run on hardware
//
#include <hip/hip_runtime.h>
#include <hip/hip_bf16.h>
#include <stdint.h>

typedef __attribute__((ext_vector_type(16))) __bf16 v16bf;
typedef __attribute__((ext_vector_type(8)))  float  v8f;

union FragB16 { v16bf v; uint4 q[2]; };

#if defined(__has_builtin) && __has_builtin(__builtin_amdgcn_sched_barrier)
#define SCHED_FENCE() __builtin_amdgcn_sched_barrier(0)
#else
#define SCHED_FENCE()
#endif

__device__ __forceinline__ float sigmoidf_(float x) { return 1.f / (1.f + __expf(-x)); }

// ---------------- f32 -> bf16 elementwise ----------------
__global__ void k_f32_to_bf16(const float* __restrict__ in, __bf16* __restrict__ out, size_t n) {
  size_t i = (size_t)blockIdx.x * blockDim.x + threadIdx.x;
  if (i < n) out[i] = (__bf16)in[i];
}

// ---------------- pack W[K,N] f32 (row major) into WMMA B-fragment order ----------------
// Layout: [K/32 ksteps][N/16 tiles][32 lanes][16 bf16] -> one kstep's tile-chunk contiguous.
// B 32x16 fragment (ISA 7.12.2): lanes 0-15 -> N=lane, K elems 0..7,16..23 of k-block;
//                                lanes 16-31 -> N=lane-16, K elems 8..15,24..31.
__global__ void k_pack_b(const float* __restrict__ W, __bf16* __restrict__ out, int K, int N) {
  size_t p = (size_t)blockIdx.x * blockDim.x + threadIdx.x;
  size_t total = (size_t)K * N;
  if (p >= total) return;
  int e    = (int)(p & 15);
  int lane = (int)((p >> 4) & 31);
  size_t rest = p >> 9;
  int ntiles = N >> 4;
  int tile = (int)(rest % ntiles);
  int ks   = (int)(rest / ntiles);
  int n = tile * 16 + (lane & 15);
  int k = ks * 32 + ((e < 8) ? ((lane >> 4) * 8 + e) : (16 + (lane >> 4) * 8 + (e - 8)));
  out[p] = (__bf16)W[(size_t)k * N + n];
}

// ---------------- GEMM: C[M,N] f32 = A[M,K] bf16 (row major) x Bpack + bias ----------------
// block = 256 (8 waves). Per-kstep 8KB B chunk double-buffered in LDS (global loads fly
// during the 8 WMMAs). B fragments: ring-of-3 register pipeline, pinned with sched_barrier
// so the scheduler cannot sink the ds_loads to their uses (partial s_wait_dscnt).
// A fragments: true ping-pong fa[ks&1] with unroll-2 (no v_mov copies).
__global__ void __launch_bounds__(256)
k_gemm_xproj(const __bf16* __restrict__ A, const __bf16* __restrict__ Bp,
             const float* __restrict__ bias, float* __restrict__ C,
             int M, int N, int K) {
  __shared__ __align__(16) __bf16 Bsh[2][8 * 512];  // 2 x 8KB
  const int tidb = threadIdx.x;
  const int lane = tidb & 31, wave = tidb >> 5, half = lane >> 4, mr = lane & 15;
  const int ksteps = K >> 5;
  const int ntiles = N >> 4;
  const int row0 = blockIdx.x * 128 + wave * 16;
  const __bf16* arow = A + (size_t)(row0 + mr) * K + half * 8;
  const int nchunks = N >> 7;
  for (int nc = 0; nc < nchunks; ++nc) {
    // ---- prologue: stage B(ks=0), load A(ks=0) ----
    const uint4* src0 = (const uint4*)(Bp + ((size_t)0 * ntiles + nc * 8) * 512);
    uint4 tb0 = src0[tidb];
    uint4 tb1 = src0[tidb + 256];
    FragB16 fa[2];
    fa[0].q[0] = *(const uint4*)(arow);
    fa[0].q[1] = *(const uint4*)(arow + 16);
    ((uint4*)Bsh[0])[tidb] = tb0;
    ((uint4*)Bsh[0])[tidb + 256] = tb1;
    v8f acc[8] = {};
    __syncthreads();
#pragma unroll 2
    for (int ks = 0; ks < ksteps; ++ks) {
      const int cur = ks & 1, nxt = cur ^ 1;
      if (ks + 1 < ksteps) {  // next kstep's globals fly during the WMMAs below (LOADcnt)
        const uint4* srcN = (const uint4*)(Bp + ((size_t)(ks + 1) * ntiles + nc * 8) * 512);
        tb0 = srcN[tidb];
        tb1 = srcN[tidb + 256];
        fa[nxt].q[0] = *(const uint4*)(arow + (ks + 1) * 32);
        fa[nxt].q[1] = *(const uint4*)(arow + (ks + 1) * 32 + 16);
      }
      // ring-of-3 LDS fragment pipeline (DScnt), prefetch distance 2, sched-pinned
      const __bf16* bsc = Bsh[cur] + lane * 16;
      FragB16 fbp[3];
      fbp[0].q[0] = *(const uint4*)(bsc);
      fbp[0].q[1] = *(const uint4*)(bsc + 8);
      fbp[1].q[0] = *(const uint4*)(bsc + 512);
      fbp[1].q[1] = *(const uint4*)(bsc + 512 + 8);
#pragma unroll
      for (int t = 0; t < 8; ++t) {
        if (t + 2 < 8) {
          const __bf16* bp = bsc + (t + 2) * 512;
          fbp[(t + 2) % 3].q[0] = *(const uint4*)(bp);
          fbp[(t + 2) % 3].q[1] = *(const uint4*)(bp + 8);
        }
        SCHED_FENCE();  // keep prefetch(t+2) above wmma(t): forces live ring + partial waits
        acc[t] = __builtin_amdgcn_wmma_f32_16x16x32_bf16(
            false, fa[cur].v, false, fbp[t % 3].v, (short)0, acc[t], false, false);
        SCHED_FENCE();
      }
      if (ks + 1 < ksteps) {
        ((uint4*)Bsh[nxt])[tidb] = tb0;
        ((uint4*)Bsh[nxt])[tidb + 256] = tb1;
      }
      __syncthreads();
    }
    // ---- epilogue ----
#pragma unroll
    for (int t = 0; t < 8; ++t) {
      int col = nc * 128 + t * 16 + mr;
      float bv = bias[col];
      float* cp = C + (size_t)(row0 + half * 8) * N + col;
#pragma unroll
      for (int r = 0; r < 8; ++r) cp[(size_t)r * N] = acc[t][r] + bv;
    }
  }
}

// ---------------- fused recurrent LSTM scan (4 instances via blockIdx.y) ----------------
// block = 512 threads (16 waves), 16 sequences, loops all T steps locally.
// Per step: G[16,2048] = h_bf16[16,512] @ WhPack; each wave owns a 128-col strip.
// Wh fragments (global/L2, LOADcnt): ring-of-4, prefetch distance 3, sched-pinned;
// wrap preloads fragments 0..2 for the next step (128 % 4 == 0 keeps ring phase).
// A fragments come from LDS (DScnt) -> independent counter, no pipeline drain.
#define H_STR 520  // padded bf16 row stride -> conflict-free b128 LDS reads

__global__ void __launch_bounds__(512)
k_lstm_scan(const float* __restrict__ xg0, const float* __restrict__ xg1,
            const float* __restrict__ xg2, const float* __restrict__ xg3,
            const __bf16* __restrict__ p0, const __bf16* __restrict__ p1,
            const __bf16* __restrict__ p2, const __bf16* __restrict__ p3,
            float* __restrict__ feats) {
  __shared__ float  G_sh[16 * 2048];   // 128 KB gate pre-activations
  __shared__ __align__(16) __bf16 h_sh[16 * H_STR];

  const int inst = blockIdx.y;
  const float*  xg = inst == 0 ? xg0 : inst == 1 ? xg1 : inst == 2 ? xg2 : xg3;
  const __bf16* Wp = inst == 0 ? p0  : inst == 1 ? p1  : inst == 2 ? p2  : p3;
  const int T = (inst < 2) ? 256 : 64;
  const int rev = inst & 1;
  const int col_off = inst * 512;

  const int tid = threadIdx.x;
  const int lane = tid & 31, wave = tid >> 5, half = lane >> 4, mr = lane & 15;
  const int seq0 = blockIdx.x * 16;

  for (int i = tid; i < 16 * H_STR; i += 512) h_sh[i] = (__bf16)0.f;
  float creg[16];
#pragma unroll
  for (int j = 0; j < 16; ++j) creg[j] = 0.f;

  // Wh fragment address for flattened index f (0..127): tile = wave*8 + (f&7), ks = f>>3.
  auto fragPtr = [&](int f) -> const __bf16* {
    return Wp + ((size_t)(f >> 3) * 128 + wave * 8 + (f & 7)) * 512 + lane * 16;
  };

  FragB16 fb[4];
#pragma unroll
  for (int i = 0; i < 3; ++i) {
    const __bf16* bp = fragPtr(i);
    fb[i].q[0] = *(const uint4*)(bp);
    fb[i].q[1] = *(const uint4*)(bp + 8);
  }
  __syncthreads();

  for (int step = 0; step < T; ++step) {
    const int t = rev ? (T - 1 - step) : step;
    // prefetch this step's xg block (16 rows x 8KB) for the elementwise phase
    {
      int l0 = tid * 2;
#pragma unroll
      for (int l = 0; l < 2; ++l) {
        int li = l0 + l;
        int j = li >> 6, off = (li & 63) << 5;
        __builtin_prefetch(xg + ((size_t)(seq0 + j) * T + t) * 2048 + off, 0, 0);
      }
    }
    // ---- matmul phase ----
    v8f acc[8] = {};
    for (int ks = 0; ks < 16; ++ks) {
      FragB16 fa;  // from LDS: DScnt, independent of the global fb pipeline
      const __bf16* ap = h_sh + mr * H_STR + ks * 32 + half * 8;
      fa.q[0] = *(const uint4*)(ap);
      fa.q[1] = *(const uint4*)(ap + 16);
#pragma unroll
      for (int tt = 0; tt < 8; ++tt) {
        int i = ks * 8 + tt;
        int nIdx = i + 3;
        const __bf16* bp = fragPtr(nIdx < 128 ? nIdx : nIdx - 128);  // wrap -> next step's 0..2
        fb[nIdx & 3].q[0] = *(const uint4*)(bp);
        fb[nIdx & 3].q[1] = *(const uint4*)(bp + 8);
        SCHED_FENCE();  // keep prefetch(i+3) above wmma(i): 3-deep global pipeline
        acc[tt] = __builtin_amdgcn_wmma_f32_16x16x32_bf16(
            false, fa.v, false, fb[i & 3].v, (short)0, acc[tt], false, false);
        SCHED_FENCE();
      }
    }
#pragma unroll
    for (int tt = 0; tt < 8; ++tt) {
      int col = wave * 128 + tt * 16 + mr;
#pragma unroll
      for (int r = 0; r < 8; ++r) G_sh[(half * 8 + r) * 2048 + col] = acc[tt][r];
    }
    __syncthreads();
    // ---- elementwise phase: thread owns hidden unit `tid` for all 16 sequences ----
#pragma unroll 4
    for (int j = 0; j < 16; ++j) {
      const float* xr = xg + ((size_t)(seq0 + j) * T + t) * 2048;
      float gi = xr[tid]        + G_sh[j * 2048 + tid];
      float gf = xr[512 + tid]  + G_sh[j * 2048 + 512 + tid];
      float gg = xr[1024 + tid] + G_sh[j * 2048 + 1024 + tid];
      float go = xr[1536 + tid] + G_sh[j * 2048 + 1536 + tid];
      float c = sigmoidf_(gf) * creg[j] + sigmoidf_(gi) * tanhf(gg);
      creg[j] = c;
      float h = sigmoidf_(go) * tanhf(c);
      h_sh[j * H_STR + tid] = (__bf16)h;
      if (step == T - 1)
        feats[(size_t)(seq0 + j) * 2048 + col_off + tid] = h;
    }
    __syncthreads();
  }
}

// ---------------- MLP: score = W2 . relu(feats_row @ W1 + b1) + b2 ----------------
__global__ void __launch_bounds__(256)
k_mlp(const float* __restrict__ feats,
      const float* __restrict__ W1a, const float* __restrict__ b1a,
      const float* __restrict__ W2a, const float* __restrict__ b2a,
      const float* __restrict__ W1b, const float* __restrict__ b1b,
      const float* __restrict__ W2b, const float* __restrict__ b2b,
      float* __restrict__ scores) {
  __shared__ float frow[2048];
  __shared__ float red[256];
  const int row = blockIdx.x;
  const int mlp = blockIdx.y;
  const float* W1 = mlp ? W1b : W1a;
  const float* b1 = mlp ? b1b : b1a;
  const float* W2 = mlp ? W2b : W2a;
  const float  b2 = (mlp ? b2b : b2a)[0];
  for (int i = threadIdx.x; i < 2048; i += 256) frow[i] = feats[(size_t)row * 2048 + i];
  __syncthreads();
  float part = 0.f;
  for (int u = threadIdx.x; u < 512; u += 256) {
    float acc = b1[u];
    for (int k = 0; k < 2048; ++k) acc += frow[k] * W1[(size_t)k * 512 + u];
    part += fmaxf(acc, 0.f) * W2[u];
  }
  red[threadIdx.x] = part;
  __syncthreads();
  for (int s = 128; s > 0; s >>= 1) {
    if (threadIdx.x < s) red[threadIdx.x] += red[threadIdx.x + s];
    __syncthreads();
  }
  if (threadIdx.x == 0) scores[mlp * 128 + row] = red[0] + b2;
}

// ---------------- select by label, mean over S, sigmoid ----------------
__global__ void k_select(const float* __restrict__ scores, const int* __restrict__ labels,
                         float* __restrict__ out) {
  int b = threadIdx.x;
  if (b >= 16) return;
  float acc = 0.f;
  for (int s = 0; s < 8; ++s) {
    int lab = labels[b * 8 + s];
    float v = (lab <= 3) ? scores[b * 8 + s] : scores[128 + b * 8 + s];
    acc += v;
  }
  out[b] = sigmoidf_(acc * 0.125f);
}

extern "C" void kernel_launch(void* const* d_in, const int* in_sizes, int n_in,
                              void* d_out, int out_size, void* d_ws, size_t ws_size,
                              hipStream_t stream) {
  // ---- input pointers (setup_inputs dict order) ----
  const float* vid_feats = (const float*)d_in[0];   // [16,8,256,1024]
  const float* txt_feats = (const float*)d_in[1];   // [16,8,64,512]
  const int*   labels    = (const int*)d_in[2];     // [16,8]
  const float* vWxf = (const float*)d_in[3];
  const float* vWhf = (const float*)d_in[4];
  const float* vbf  = (const float*)d_in[5];
  const float* vWxb = (const float*)d_in[6];
  const float* vWhb = (const float*)d_in[7];
  const float* vbb  = (const float*)d_in[8];
  const float* tWxf = (const float*)d_in[9];
  const float* tWhf = (const float*)d_in[10];
  const float* tbf  = (const float*)d_in[11];
  const float* tWxb = (const float*)d_in[12];
  const float* tWhb = (const float*)d_in[13];
  const float* tbb  = (const float*)d_in[14];
  const float* sqW1 = (const float*)d_in[15];
  const float* sqb1 = (const float*)d_in[16];
  const float* sqW2 = (const float*)d_in[17];
  const float* sqb2 = (const float*)d_in[18];
  const float* rqW1 = (const float*)d_in[19];
  const float* rqb1 = (const float*)d_in[20];
  const float* rqW2 = (const float*)d_in[21];
  const float* rqb2 = (const float*)d_in[22];

  const size_t MV = 128 * 256, KV = 1024;
  const size_t MT = 128 * 64,  KT = 512;
  const size_t G4H = 2048;

  // ---- carve workspace ----
  char* w = (char*)d_ws;
  auto carve = [&](size_t bytes) -> void* {
    void* p = (void*)w;
    w += (bytes + 255) & ~(size_t)255;
    return p;
  };
  float*  xg_vf = (float*)carve(MV * G4H * 4);
  float*  xg_vb = (float*)carve(MV * G4H * 4);
  float*  xg_tf = (float*)carve(MT * G4H * 4);
  float*  xg_tb = (float*)carve(MT * G4H * 4);
  __bf16* Avid  = (__bf16*)carve(MV * KV * 2);
  __bf16* Atxt  = (__bf16*)carve(MT * KT * 2);
  __bf16* pvWxf = (__bf16*)carve(KV * G4H * 2);
  __bf16* pvWxb = (__bf16*)carve(KV * G4H * 2);
  __bf16* pvWhf = (__bf16*)carve(512 * G4H * 2);
  __bf16* pvWhb = (__bf16*)carve(512 * G4H * 2);
  __bf16* ptWxf = (__bf16*)carve(KT * G4H * 2);
  __bf16* ptWxb = (__bf16*)carve(KT * G4H * 2);
  __bf16* ptWhf = (__bf16*)carve(512 * G4H * 2);
  __bf16* ptWhb = (__bf16*)carve(512 * G4H * 2);
  float*  feats  = (float*)carve(128 * G4H * 4);
  float*  scores = (float*)carve(256 * 4);

  // ---- 1. activation conversion to bf16 ----
  {
    size_t n = MV * KV;
    k_f32_to_bf16<<<(unsigned)((n + 255) / 256), 256, 0, stream>>>(vid_feats, Avid, n);
    n = MT * KT;
    k_f32_to_bf16<<<(unsigned)((n + 255) / 256), 256, 0, stream>>>(txt_feats, Atxt, n);
  }
  // ---- 2. weight packing into WMMA fragment order ----
  auto pack = [&](const float* W, __bf16* out, int K, int N) {
    size_t n = (size_t)K * N;
    k_pack_b<<<(unsigned)((n + 255) / 256), 256, 0, stream>>>(W, out, K, N);
  };
  pack(vWxf, pvWxf, 1024, 2048);
  pack(vWxb, pvWxb, 1024, 2048);
  pack(vWhf, pvWhf, 512, 2048);
  pack(vWhb, pvWhb, 512, 2048);
  pack(tWxf, ptWxf, 512, 2048);
  pack(tWxb, ptWxb, 512, 2048);
  pack(tWhf, ptWhf, 512, 2048);
  pack(tWhb, ptWhb, 512, 2048);

  // ---- 3. input projections xg = x @ Wx + b ----
  k_gemm_xproj<<<(unsigned)(MV / 128), 256, 0, stream>>>(Avid, pvWxf, vbf, xg_vf, (int)MV, 2048, 1024);
  k_gemm_xproj<<<(unsigned)(MV / 128), 256, 0, stream>>>(Avid, pvWxb, vbb, xg_vb, (int)MV, 2048, 1024);
  k_gemm_xproj<<<(unsigned)(MT / 128), 256, 0, stream>>>(Atxt, ptWxf, tbf, xg_tf, (int)MT, 2048, 512);
  k_gemm_xproj<<<(unsigned)(MT / 128), 256, 0, stream>>>(Atxt, ptWxb, tbb, xg_tb, (int)MT, 2048, 512);

  // ---- 4. fused recurrent scans ----
  {
    dim3 grid(8, 4);
    k_lstm_scan<<<grid, 512, 0, stream>>>(xg_vf, xg_vb, xg_tf, xg_tb,
                                          pvWhf, pvWhb, ptWhf, ptWhb, feats);
  }
  // ---- 5. MLP heads + selection epilogue ----
  {
    dim3 grid(128, 2);
    k_mlp<<<grid, 256, 0, stream>>>(feats, sqW1, sqb1, sqW2, sqb2,
                                    rqW1, rqb1, rqW2, rqb2, scores);
  }
  k_select<<<1, 32, 0, stream>>>(scores, labels, (float*)d_out);
}
